// MultiHeadAttention_76673756168890
// MI455X (gfx1250) — matrix operand order repaired and checked by reading, not compiled, not run on hardware
//
#include <hip/hip_runtime.h>

typedef __attribute__((ext_vector_type(16))) __bf16        v16bf;
typedef __attribute__((ext_vector_type(2)))  __bf16        v2bf;
typedef __attribute__((ext_vector_type(8)))  float         v8f;
typedef __attribute__((ext_vector_type(8)))  unsigned int  v8u;
typedef __attribute__((ext_vector_type(4)))  unsigned int  v4u;
typedef __attribute__((ext_vector_type(2)))  unsigned int  v2u;
typedef __attribute__((ext_vector_type(4)))  int           v4i;

#if defined(__gfx1250__) && \
    __has_builtin(__builtin_amdgcn_global_load_async_to_lds_b128) && \
    __has_builtin(__builtin_amdgcn_s_wait_asynccnt)
#define HAVE_ASYNC_LDS 1
#else
#define HAVE_ASYNC_LDS 0
#endif

static __device__ __forceinline__ __bf16 f2bf(float f) {
  unsigned u = __builtin_bit_cast(unsigned, f);
  unsigned r = u + 0x7FFFu + ((u >> 16) & 1u);   // round-to-nearest-even
  unsigned short h = (unsigned short)(r >> 16);
  return __builtin_bit_cast(__bf16, h);
}

// pack two f32 -> dword of two bf16 (hardware packed convert when available)
static __device__ __forceinline__ unsigned pack2bf(float a, float b) {
#if defined(__gfx1250__) && __has_builtin(__builtin_amdgcn_cvt_pk_bf16_f32)
  const v2bf p = __builtin_amdgcn_cvt_pk_bf16_f32(a, b);
  return __builtin_bit_cast(unsigned, p);
#else
  return (unsigned)__builtin_bit_cast(unsigned short, f2bf(a)) |
         ((unsigned)__builtin_bit_cast(unsigned short, f2bf(b)) << 16);
#endif
}

static __device__ __forceinline__ v8f vzero8() {
  v8f z;
#pragma unroll
  for (int i = 0; i < 8; ++i) z[i] = 0.0f;
  return z;
}

static __device__ __forceinline__ v8f wmma_bf16(v16bf a, v16bf b, v8f c) {
  return __builtin_amdgcn_wmma_f32_16x16x32_bf16(false, a, false, b, (short)0, c,
                                                 false, false);
}

// two 16-byte LDS reads -> one 16-element bf16 fragment
static __device__ __forceinline__ v16bf ldfrag2(const __bf16* p0, const __bf16* p1) {
  const v4u a = *(const v4u*)p0;
  const v4u b = *(const v4u*)p1;
  v8u u;
#pragma unroll
  for (int j = 0; j < 4; ++j) { u[j] = a[j]; u[4 + j] = b[j]; }
  return __builtin_bit_cast(v16bf, u);
}

// 16-byte global->LDS copy: async on gfx1250, sync fallback otherwise
static __device__ __forceinline__ void cp16_g2l(const __bf16* gsrc, __bf16* ldst) {
#if HAVE_ASYNC_LDS
  typedef __attribute__((address_space(1))) v4i GV4;   // global int4
  typedef __attribute__((address_space(3))) v4i LV4;   // LDS int4
  __builtin_amdgcn_global_load_async_to_lds_b128((GV4*)gsrc, (LV4*)ldst, 0, 0);
#else
  *(v4u*)ldst = *(const v4u*)gsrc;
#endif
}

static __device__ __forceinline__ void async_wait_all() {
#if HAVE_ASYNC_LDS
  __builtin_amdgcn_s_wait_asynccnt(0);
#endif
}

#define UD 1024

// ---------------------------------------------------------------------------
// Kernel 1: fused QKVR GEMM  out = relu(x @ W + b)
// q/k/v written as bf16 (consumed by WMMA attention), res written as f32.
// grid = (64 Mtiles, 8 Ntiles, 4 weights), block = 256 (8 waves)
// ---------------------------------------------------------------------------
__global__ __launch_bounds__(256) void qkvr_gemm(
    const float* __restrict__ x,
    const float* __restrict__ Wq, const float* __restrict__ bq,
    const float* __restrict__ Wk, const float* __restrict__ bk,
    const float* __restrict__ Wv, const float* __restrict__ bv,
    const float* __restrict__ Wr, const float* __restrict__ br,
    __bf16* __restrict__ oq, __bf16* __restrict__ ok,
    __bf16* __restrict__ ov, float* __restrict__ orr)
{
  const float* W; const float* bias;
  __bf16* outB = nullptr; float* outF = nullptr;
  switch (blockIdx.z) {
    case 0:  W = Wq; bias = bq; outB = oq;  break;
    case 1:  W = Wk; bias = bk; outB = ok;  break;
    case 2:  W = Wv; bias = bv; outB = ov;  break;
    default: W = Wr; bias = br; outF = orr; break;
  }

  __shared__ __align__(16) __bf16 As[128][40];   // x tile, row-major (M x K)
  __shared__ __align__(16) __bf16 BsT[128][40];  // W tile, transposed (N x K)

  const int tid  = threadIdx.x;
  const int lane = tid & 31;
  const int wv   = tid >> 5;    // 0..7
  const int wm   = wv & 3;      // 32-row group
  const int wn   = wv >> 2;     // 64-col group
  const int half = lane >> 4;   // 0/1
  const int lr   = lane & 15;

  const int mBase = blockIdx.x * 128;
  const int nBase = blockIdx.y * 128;

  v8f acc[2][4];
#pragma unroll
  for (int i = 0; i < 2; ++i)
#pragma unroll
    for (int j = 0; j < 4; ++j) acc[i][j] = vzero8();

  for (int kc = 0; kc < UD; kc += 32) {
    // stage x tile (128 x 32): float4 -> packed bf16 pair dwords -> b64 store
#pragma unroll
    for (int i = 0; i < 4; ++i) {
      const int idx = tid + 256 * i;   // 1024 float4s
      const int row = idx >> 3;
      const int c4  = idx & 7;
      const float4 f =
          *(const float4*)(x + (size_t)(mBase + row) * UD + kc + 4 * c4);
      v2u t;
      t[0] = pack2bf(f.x, f.y);
      t[1] = pack2bf(f.z, f.w);
      *(v2u*)&As[row][4 * c4] = t;
    }
    // stage W tile (32 x 128) transposed: read 8 k at fixed n (coalesced b32
    // across lanes), pack pairs, one b128 store per 8 elements
#pragma unroll
    for (int t2 = 0; t2 < 2; ++t2) {
      const int tile = tid + 256 * t2;   // 512 tiles = 128 n x 4 k-octets
      const int n  = tile & 127;
      const int kq = tile >> 7;
      float w[8];
#pragma unroll
      for (int j = 0; j < 8; ++j)
        w[j] = W[(size_t)(kc + kq * 8 + j) * UD + nBase + n];
      v4u t;
#pragma unroll
      for (int j = 0; j < 4; ++j) t[j] = pack2bf(w[2 * j], w[2 * j + 1]);
      *(v4u*)&BsT[n][kq * 8] = t;
    }
    __syncthreads();

    // A fragments: ISA 16-bit A 16x32 layout (K halves 0..7 / 8..15, +16)
    v16bf afr[2];
#pragma unroll
    for (int mi = 0; mi < 2; ++mi) {
      const int row = 32 * wm + 16 * mi + lr;
      afr[mi] = ldfrag2(&As[row][half * 8], &As[row][16 + half * 8]);
    }
#pragma unroll
    for (int ni = 0; ni < 4; ++ni) {
      // B fragment: lane holds column, K halves 0..7 / 8..15, +16 (matches A)
      const int col = 64 * wn + 16 * ni + lr;
      const v16bf bfr = ldfrag2(&BsT[col][half * 8], &BsT[col][16 + half * 8]);
#pragma unroll
      for (int mi = 0; mi < 2; ++mi)
        acc[mi][ni] = wmma_bf16(afr[mi], bfr, acc[mi][ni]);
    }
    __syncthreads();
  }

  // epilogue: bias + relu (C layout: VGPR r -> M = r + 8*half, N = lr)
  // uniform branch hoisted outside the store loops
  if (outF) {
#pragma unroll
    for (int mi = 0; mi < 2; ++mi)
#pragma unroll
      for (int ni = 0; ni < 4; ++ni) {
        const int gcol = nBase + 64 * wn + 16 * ni + lr;
        const float b  = bias[gcol];
#pragma unroll
        for (int r = 0; r < 8; ++r) {
          const int grow = mBase + 32 * wm + 16 * mi + r + 8 * half;
          outF[(size_t)grow * UD + gcol] = fmaxf(acc[mi][ni][r] + b, 0.0f);
        }
      }
  } else {
#pragma unroll
    for (int mi = 0; mi < 2; ++mi)
#pragma unroll
      for (int ni = 0; ni < 4; ++ni) {
        const int gcol = nBase + 64 * wn + 16 * ni + lr;
        const float b  = bias[gcol];
#pragma unroll
        for (int r = 0; r < 8; ++r) {
          const int grow = mBase + 32 * wm + 16 * mi + r + 8 * half;
          outB[(size_t)grow * UD + gcol] =
              f2bf(fmaxf(acc[mi][ni][r] + b, 0.0f));
        }
      }
  }
}

// ---------------------------------------------------------------------------
// Kernel 2: flash attention over 32 head matrices of (2048 x 128), bf16 in
// grid = (32 query blocks of 64 rows, 32 heads), block = 128 (4 waves)
// ---------------------------------------------------------------------------
__global__ __launch_bounds__(128) void flash_attn(
    const __bf16* __restrict__ qg, const __bf16* __restrict__ kg,
    const __bf16* __restrict__ vg, float* __restrict__ og)
{
  const int head = blockIdx.y;
  const int qb   = blockIdx.x;
  const size_t base = (size_t)head * 2048 * 128;

  const int tid  = threadIdx.x;
  const int lane = tid & 31;
  const int wv   = tid >> 5;    // 0..3, each wave owns 16 query rows
  const int half = lane >> 4;
  const int lr   = lane & 15;

  __shared__ __align__(16) __bf16 Qs[64][136];   // Q block (rows x HD)
  __shared__ __align__(16) __bf16 Ks[32][136];   // K block (keys x HD)
  __shared__ __align__(16) __bf16 VsT[128][40];  // V block transposed (HD x keys)
  __shared__ __align__(16) float  Ss[4][16][32]; // per-wave raw scores
  __shared__ __align__(16) __bf16 Ps[4][16][32]; // per-wave softmax'd P (bf16)
  __shared__ float  rs[4][16];                   // per-row scale broadcast

  // stage Q block (64 x 128 bf16): 1024 chunks of 16B, async copy
#pragma unroll
  for (int i = 0; i < 8; ++i) {
    const int idx = tid + 128 * i;
    const int row = idx >> 4;           // 16 chunks per row
    const int c   = idx & 15;
    cp16_g2l(qg + base + (size_t)(qb * 64 + row) * 128 + c * 8, &Qs[row][c * 8]);
  }

  v8f o[8];
#pragma unroll
  for (int f = 0; f < 8; ++f) o[f] = vzero8();
  float m = -1e30f, l = 0.0f;
  const float scale = 0.08838834764831845f;  // 1/sqrt(128)

  for (int kb = 0; kb < 2048; kb += 32) {
    __syncthreads();  // previous iteration done reading Ks/VsT

    // stage K block (32 x 128 bf16): 512 chunks of 16B, async copy
#pragma unroll
    for (int i = 0; i < 4; ++i) {
      const int idx = tid + 128 * i;
      const int row = idx >> 4;
      const int c   = idx & 15;
      cp16_g2l(kg + base + (size_t)(kb + row) * 128 + c * 8, &Ks[row][c * 8]);
    }
    // stage V transposed: per tile read 8 keys x 2 hd, pack, two b128 stores
#pragma unroll
    for (int t2 = 0; t2 < 2; ++t2) {
      const int tile = tid + 128 * t2;    // 0..255
      const int hp   = tile & 63;         // hd pair (coalesced across lanes)
      const int kq   = tile >> 6;         // key octet
      unsigned u[8];
#pragma unroll
      for (int j = 0; j < 8; ++j)
        u[j] = *(const unsigned*)(vg + base +
                                  (size_t)(kb + kq * 8 + j) * 128 + 2 * hp);
      v4u lo, hi;
#pragma unroll
      for (int j = 0; j < 4; ++j) {
        lo[j] = (u[2 * j] & 0xFFFFu) | (u[2 * j + 1] << 16);
        hi[j] = (u[2 * j] >> 16) | (u[2 * j + 1] & 0xFFFF0000u);
      }
      *(v4u*)&VsT[2 * hp + 0][kq * 8] = lo;
      *(v4u*)&VsT[2 * hp + 1][kq * 8] = hi;
    }
    async_wait_all();
    __syncthreads();

    // S = Q K^T : this wave's 16 rows x 32 keys, K-depth 128 in 4 WMMA steps
    v8f s0 = vzero8(), s1 = vzero8();
#pragma unroll
    for (int kk = 0; kk < 4; ++kk) {
      const int row = wv * 16 + lr;
      const v16bf a = ldfrag2(&Qs[row][kk * 32 + half * 8],
                              &Qs[row][kk * 32 + 16 + half * 8]);
#pragma unroll
      for (int ni = 0; ni < 2; ++ni) {
        const int key = ni * 16 + lr;     // B column = key index
        const v16bf b = ldfrag2(&Ks[key][kk * 32 + half * 8],
                                &Ks[key][kk * 32 + 16 + half * 8]);
        if (ni == 0) s0 = wmma_bf16(a, b, s0);
        else         s1 = wmma_bf16(a, b, s1);
      }
    }
    // spill scaled scores to this wave's LDS region
#pragma unroll
    for (int r = 0; r < 8; ++r) {
      Ss[wv][r + 8 * half][lr]      = s0[r] * scale;
      Ss[wv][r + 8 * half][16 + lr] = s1[r] * scale;
    }
    __builtin_amdgcn_wave_barrier();

    // online softmax: lanes (t, t+16) co-own row t; each handles 16 columns
    float vals[16];
    float cmax = -1e30f;
#pragma unroll
    for (int c = 0; c < 16; ++c) {
      vals[c] = Ss[wv][lr][half * 16 + c];
      cmax = fmaxf(cmax, vals[c]);
    }
    cmax = fmaxf(cmax, __shfl_xor(cmax, 16, 32));
    const float newm = fmaxf(m, cmax);
    float p[16];
    float psum = 0.0f;
#pragma unroll
    for (int c = 0; c < 16; ++c) {
      p[c] = __expf(vals[c] - newm);
      psum += p[c];
    }
    // packed bf16 store of P: 16 contiguous elements -> two b128 stores
    {
      v4u pk0, pk1;
#pragma unroll
      for (int j = 0; j < 4; ++j) {
        pk0[j] = pack2bf(p[2 * j],     p[2 * j + 1]);
        pk1[j] = pack2bf(p[8 + 2 * j], p[8 + 2 * j + 1]);
      }
      *(v4u*)&Ps[wv][lr][half * 16]     = pk0;
      *(v4u*)&Ps[wv][lr][half * 16 + 8] = pk1;
    }
    psum += __shfl_xor(psum, 16, 32);
    const float alpha = __expf(m - newm);
    l = l * alpha + psum;
    m = newm;
    if (half == 0) rs[wv][lr] = alpha;
    __builtin_amdgcn_wave_barrier();

    // rescale running O by alpha (per-row broadcast via LDS)
#pragma unroll
    for (int f = 0; f < 8; ++f)
#pragma unroll
      for (int r = 0; r < 8; ++r) o[f][r] *= rs[wv][r + 8 * half];

    // O += P V : A = P (16x32 bf16), B = V^T columns (contiguous K runs)
    {
      const v16bf a = ldfrag2(&Ps[wv][lr][half * 8],
                              &Ps[wv][lr][16 + half * 8]);
#pragma unroll
      for (int f = 0; f < 8; ++f) {
        const int col = f * 16 + lr;  // output HD column
        const v16bf b = ldfrag2(&VsT[col][half * 8], &VsT[col][16 + half * 8]);
        o[f] = wmma_bf16(a, b, o[f]);
      }
    }
  }

  // finalize: O /= l, write out (f32)
  if (half == 0) rs[wv][lr] = 1.0f / l;
  __builtin_amdgcn_wave_barrier();
#pragma unroll
  for (int f = 0; f < 8; ++f)
#pragma unroll
    for (int r = 0; r < 8; ++r) {
      const int grow = qb * 64 + wv * 16 + r + 8 * half;
      og[base + (size_t)grow * 128 + f * 16 + lr] =
          o[f][r] * rs[wv][r + 8 * half];
    }
}

// ---------------------------------------------------------------------------
// Kernel 3: out = LayerNorm(relu(attn + res)) * gamma + beta
// grid = 8192 rows, block = 256
// ---------------------------------------------------------------------------
__global__ __launch_bounds__(256) void relu_ln(
    const float* __restrict__ attn, const float* __restrict__ res,
    const float* __restrict__ gamma, const float* __restrict__ beta,
    float* __restrict__ out)
{
  const int row = blockIdx.x;
  const int tid = threadIdx.x;
  const float* a = attn + (size_t)row * 1024;
  const float* r = res  + (size_t)row * 1024;

  float y[4];
  float s = 0.0f, s2 = 0.0f;
#pragma unroll
  for (int i = 0; i < 4; ++i) {
    const int idx = tid + 256 * i;
    const float v = fmaxf(a[idx] + r[idx], 0.0f);
    y[i] = v; s += v; s2 += v * v;
  }
#pragma unroll
  for (int off = 16; off > 0; off >>= 1) {
    s  += __shfl_xor(s,  off, 32);
    s2 += __shfl_xor(s2, off, 32);
  }
  __shared__ float red[8], red2[8];
  const int wv = tid >> 5;
  if ((tid & 31) == 0) { red[wv] = s; red2[wv] = s2; }
  __syncthreads();
  float ts = 0.0f, ts2 = 0.0f;
#pragma unroll
  for (int i = 0; i < 8; ++i) { ts += red[i]; ts2 += red2[i]; }
  const float mean = ts * (1.0f / 1024.0f);
  const float var  = ts2 * (1.0f / 1024.0f) - mean * mean;
  const float rinv = rsqrtf(var + 1e-8f);
#pragma unroll
  for (int i = 0; i < 4; ++i) {
    const int idx = tid + 256 * i;
    out[(size_t)row * 1024 + idx] = gamma[idx] * (y[i] - mean) * rinv + beta[idx];
  }
}

// ---------------------------------------------------------------------------
extern "C" void kernel_launch(void* const* d_in, const int* in_sizes, int n_in,
                              void* d_out, int out_size, void* d_ws, size_t ws_size,
                              hipStream_t stream) {
  const float* x  = (const float*)d_in[0];
  const float* Wq = (const float*)d_in[1];
  const float* bq = (const float*)d_in[2];
  const float* Wk = (const float*)d_in[3];
  const float* bk = (const float*)d_in[4];
  const float* Wv = (const float*)d_in[5];
  const float* bv = (const float*)d_in[6];
  const float* Wr = (const float*)d_in[7];
  const float* br = (const float*)d_in[8];
  const float* gamma = (const float*)d_in[9];
  const float* beta  = (const float*)d_in[10];

  const size_t N = (size_t)8192 * 1024;   // elements per activation buffer
  __bf16* q = (__bf16*)d_ws;              // bf16 activations for attention
  __bf16* k = q + N;
  __bf16* v = k + N;
  float*  r    = (float*)(v + N);         // residual kept in f32
  float*  attn = r + N;

  qkvr_gemm<<<dim3(64, 8, 4), 256, 0, stream>>>(x, Wq, bq, Wk, bk, Wv, bv,
                                                Wr, br, q, k, v, r);
  flash_attn<<<dim3(32, 32), 128, 0, stream>>>(q, k, v, attn);
  relu_ln<<<8192, 256, 0, stream>>>(attn, r, gamma, beta, (float*)d_out);
}
